// SRCMBlock_5145370821189
// MI455X (gfx1250) — matrix operand order repaired and checked by reading, not compiled, not run on hardware
//
#include <hip/hip_runtime.h>
#include <hip/hip_bf16.h>
#include <math.h>

// ---------------- problem constants ----------------
#define BB 4
#define CC 256
#define HHH 48
#define WWW 48
#define LL (HHH*WWW)          // 2304
#define GG 4
#define DG 64
#define DI 128
#define NST 16
#define RR 4
#define KCONV 4
#define MROWS (BB*LL)         // 9216
#define DBLN (RR+2*NST)       // 36
#define DBLP 48               // padded leading dim for dbl
#define EPSV 1e-5f

typedef __attribute__((ext_vector_type(16))) __bf16 v16bf;
typedef __attribute__((ext_vector_type(8)))  float  v8f;

__device__ __forceinline__ unsigned short bfbits(float f) {
    union { float f; unsigned u; } x; x.f = f;
    unsigned r = x.u + 0x7FFFu + ((x.u >> 16) & 1u);   // round-to-nearest-even
    return (unsigned short)(r >> 16);
}
__device__ __forceinline__ unsigned pk2(float lo, float hi) {
#if __has_builtin(__builtin_amdgcn_cvt_pk_bf16_f32)
    typedef __attribute__((ext_vector_type(2))) __bf16 v2bf;
    v2bf p = __builtin_amdgcn_cvt_pk_bf16_f32(lo, hi);
    return __builtin_bit_cast(unsigned, p);
#else
    return (unsigned)bfbits(lo) | ((unsigned)bfbits(hi) << 16);
#endif
}
__device__ __forceinline__ float sigmf(float v) { return 1.f / (1.f + __expf(-v)); }
__device__ __forceinline__ float siluf(float v) { return v * sigmf(v); }

// ---------------- generic WMMA bf16 GEMM: Y = act(X @ W^T + bias) ----------------
// Requirements (all satisfied by this pipeline): M % 32 == 0, K % 64 == 0,
// X rows 16-byte aligned (ldx, xoff multiples of 4 floats).
// Block tile: 32 rows x 64 cols, K-step 64, 8 waves, 2 WMMA / wave / iter.
__global__ __launch_bounds__(256) void gemm_wmma_bf16(
    const float* __restrict__ X, int ldx, int xoff,
    const float* __restrict__ W,        // N x K row-major
    const float* __restrict__ bias,
    float* __restrict__ Y, int ldy, int yoff,
    int N, int K,
    int act, int storeMode)
{
    // rows of 64 bf16 + 8 pad = 72 bf16 = 36 dwords
    __shared__ unsigned AsU[32][36];
    __shared__ unsigned BsU[64][36];
    const __bf16* As = (const __bf16*)AsU;   // row stride 72
    const __bf16* Bs = (const __bf16*)BsU;

    const int tid  = threadIdx.x;
    const int lane = tid & 31;
    const int wave = tid >> 5;            // 8 waves
    const int wm   = wave >> 2;           // 0..1
    const int wn   = wave & 3;            // 0..3
    const int rowBase = blockIdx.y * 32;
    const int colBase = blockIdx.x * 64;

    // per-thread staging coordinates (fixed across K loop)
    const int ar  = tid >> 4;             // A row 0..15 (+16 for second chunk)
    const int ac4 = (tid & 15) * 4;       // A col (float4 granularity)

    v8f acc = {0.f,0.f,0.f,0.f,0.f,0.f,0.f,0.f};

    for (int k0 = 0; k0 < K; k0 += 64) {
        // ---- issue ALL tile loads first (6x b128), convert/store after
        float4 avv[2], bvv[4];
        #pragma unroll
        for (int i = 0; i < 2; ++i) {
            int r = ar + i * 16;
            avv[i] = *(const float4*)(X + (size_t)(rowBase + r) * ldx + xoff + k0 + ac4);
        }
        #pragma unroll
        for (int i = 0; i < 4; ++i) {
            int r  = ar + i * 16;
            int gn = colBase + r;
            int rn = (gn < N) ? gn : 0;   // clamped, unconditional load
            bvv[i] = *(const float4*)(W + (size_t)rn * K + k0 + ac4);
        }
        if (k0 + 64 < K) {
            __builtin_prefetch(X + (size_t)(rowBase + wave * 4) * ldx + xoff + k0 + 64, 0, 1);
        }
        #pragma unroll
        for (int i = 0; i < 2; ++i) {
            int r = ar + i * 16;
            *(uint2*)&AsU[r][ac4 >> 1] =
                make_uint2(pk2(avv[i].x, avv[i].y), pk2(avv[i].z, avv[i].w));
        }
        #pragma unroll
        for (int i = 0; i < 4; ++i) {
            int r  = ar + i * 16;
            int gn = colBase + r;
            float4 v = bvv[i];
            if (gn >= N) { v.x = 0.f; v.y = 0.f; v.z = 0.f; v.w = 0.f; }
            *(uint2*)&BsU[r][ac4 >> 1] = make_uint2(pk2(v.x, v.y), pk2(v.z, v.w));
        }
        __syncthreads();

        const int half = (lane >> 4) & 1;
        const int sub  = lane & 15;
        v16bf a0, a1, b0, b1;
        const int arow = (wm * 16 + sub) * 72;
        const int brow = (wn * 16 + sub) * 72;
        #pragma unroll
        for (int vv = 0; vv < 8; ++vv) {
            // A 16x32 layout: lanes0-15 K={0..7,16..23}, lanes16-31 K={8..15,24..31}
            int ka = (vv >> 2) * 16 + half * 8 + 2 * (vv & 3);
            a0[2*vv]   = As[arow + ka];
            a0[2*vv+1] = As[arow + ka + 1];
            a1[2*vv]   = As[arow + 32 + ka];
            a1[2*vv+1] = As[arow + 32 + ka + 1];
            // B 32x16 layout: lanes0-15 K={0..7,16..23}, lanes16-31 K={8..15,24..31}
            int kb = (vv >> 2) * 16 + half * 8 + 2 * (vv & 3);
            b0[2*vv]   = Bs[brow + kb];
            b0[2*vv+1] = Bs[brow + kb + 1];
            b1[2*vv]   = Bs[brow + 32 + kb];
            b1[2*vv+1] = Bs[brow + 32 + kb + 1];
        }
        acc = __builtin_amdgcn_wmma_f32_16x16x32_bf16(false, a0, false, b0, (short)0, acc, false, false);
        acc = __builtin_amdgcn_wmma_f32_16x16x32_bf16(false, a1, false, b1, (short)0, acc, false, false);
        __syncthreads();
    }

    // epilogue: C/D 16x16 f32 layout (VGPR r -> row 8*half+r, lane&15 -> col)
    const int half = (lane >> 4) & 1;
    const int sub  = lane & 15;
    #pragma unroll
    for (int r = 0; r < 8; ++r) {
        int grow = rowBase + wm * 16 + half * 8 + r;
        int gcol = colBase + wn * 16 + sub;
        if (gcol < N) {
            float v = acc[r];
            if (bias) v += bias[gcol];
            if (act == 1) v = sigmf(v);
            if (storeMode == 0) {
                Y[(size_t)grow * ldy + yoff + gcol] = v;
            } else {
                int bidx = grow / LL, l = grow - bidx * LL;  // LL is compile-time
                Y[((size_t)bidx * N + gcol) * LL + l] = v;
            }
        }
    }
}

// ---------------- instance norm + relu over (H*W) per (b,c) ----------------
__global__ __launch_bounds__(256) void inorm_relu_kernel(const float* __restrict__ x,
                                                         float* __restrict__ y)
{
    int bc = blockIdx.x;
    const float* xp = x + (size_t)bc * LL;
    float s = 0.f, s2 = 0.f;
    for (int t = threadIdx.x; t < LL; t += 256) { float v = xp[t]; s += v; s2 += v*v; }
    __shared__ float r1[256], r2[256];
    r1[threadIdx.x] = s; r2[threadIdx.x] = s2; __syncthreads();
    for (int off = 128; off > 0; off >>= 1) {
        if (threadIdx.x < off) { r1[threadIdx.x] += r1[threadIdx.x+off];
                                 r2[threadIdx.x] += r2[threadIdx.x+off]; }
        __syncthreads();
    }
    float mu  = r1[0] / (float)LL;
    float var = r2[0] / (float)LL - mu * mu;
    float inv = rsqrtf(var + EPSV);
    float* yp = y + (size_t)bc * LL;
    for (int t = threadIdx.x; t < LL; t += 256)
        yp[t] = fmaxf(0.f, (xp[t] - mu) * inv);
}

// ---------------- tokens = hn(transposed) + bilinear(pos); layernorm over C ----------------
__global__ __launch_bounds__(256) void tokens_ln_kernel(
    const float* __restrict__ hn,   // (B,C,L)
    const float* __restrict__ pos,  // (64*64, C)
    const float* __restrict__ lng, const float* __restrict__ lnb,
    float* __restrict__ tokens, float* __restrict__ xn)
{
    int bl = blockIdx.x;
    int b = bl / LL, l = bl % LL;
    int i = l / WWW, j = l % WWW;
    int c = threadIdx.x;

    float fy = (i + 0.5f) * (64.f / 48.f) - 0.5f;
    float fx = (j + 0.5f) * (64.f / 48.f) - 0.5f;
    int y0 = (int)floorf(fy), x0 = (int)floorf(fx);
    float wy = fy - (float)y0, wx = fx - (float)x0;
    int y1 = y0 + 1 > 63 ? 63 : y0 + 1;
    int x1 = x0 + 1 > 63 ? 63 : x0 + 1;
    y0 = y0 < 0 ? 0 : y0; x0 = x0 < 0 ? 0 : x0;
    float p00 = pos[((size_t)y0*64 + x0)*CC + c];
    float p01 = pos[((size_t)y0*64 + x1)*CC + c];
    float p10 = pos[((size_t)y1*64 + x0)*CC + c];
    float p11 = pos[((size_t)y1*64 + x1)*CC + c];
    float pv  = p00*(1.f-wy)*(1.f-wx) + p01*(1.f-wy)*wx + p10*wy*(1.f-wx) + p11*wy*wx;

    float tv = hn[((size_t)b*CC + c)*LL + l] + pv;

    __shared__ float s1[256], s2[256];
    s1[c] = tv; s2[c] = tv*tv; __syncthreads();
    for (int off = 128; off > 0; off >>= 1) {
        if (c < off) { s1[c] += s1[c+off]; s2[c] += s2[c+off]; }
        __syncthreads();
    }
    float mu  = s1[0] / (float)CC;
    float var = s2[0] / (float)CC - mu * mu;
    float inv = rsqrtf(var + EPSV);
    size_t o = ((size_t)b*LL + l)*CC + c;
    tokens[o] = tv;
    xn[o]     = (tv - mu) * inv * lng[c] + lnb[c];
}

// ---------------- causal depthwise conv (K=4) + SiLU, fwd/bwd ----------------
__global__ __launch_bounds__(256) void conv_silu_kernel(
    const float* __restrict__ xz,   // [G][B*L][256], xm = cols 0..DI-1
    const float* __restrict__ cw,   // (G,DI,KCONV)
    const float* __restrict__ cb,   // (G,DI)
    float* __restrict__ uc,         // [G][B*L][DI] (direction-local order)
    int dir)
{
    int idx = blockIdx.x * 256 + threadIdx.x;      // (g,b,t,d), d fastest
    int d  = idx & (DI - 1);
    int t  = (idx >> 7) % LL;
    int gb = idx / (LL * DI);
    int g  = gb / BB, b = gb % BB;
    float accv = 0.f;
    #pragma unroll
    for (int k = 0; k < KCONV; ++k) {
        int ti = t - (KCONV - 1) + k;
        if (ti >= 0) {
            int sp = dir ? (LL - 1 - ti) : ti;
            accv += cw[((size_t)g*DI + d)*KCONV + k] *
                    xz[(((size_t)(g*BB + b))*LL + sp)*(2*DI) + d];
        }
    }
    accv += cb[(size_t)g*DI + d];
    uc[idx] = siluf(accv);
}

// ---------------- selective scan (one thread per channel d; 16 states in regs) ----------------
__global__ __launch_bounds__(DI) void scan_kernel(
    const float* __restrict__ uc,    // [G][B*L][DI]
    const float* __restrict__ dbl,   // [G][B*L][DBLP]: [0..3]=dt-in, [4..19]=B, [20..35]=C
    const float* __restrict__ xz,    // [G][B*L][256], z = cols DI..2*DI-1
    const float* __restrict__ dtw,   // (G,DI,RR)
    const float* __restrict__ dtb,   // (G,DI)
    const float* __restrict__ A_log, // (G,DI,NST)
    const float* __restrict__ Dp,    // (G,DI)
    float* __restrict__ ySum,        // [G][B*L][DI]
    int dir)                         // 0: store (fwd), 1: add at reversed pos (bwd)
{
    int g = blockIdx.x / BB, b = blockIdx.x % BB;
    int d = threadIdx.x;

    float w0 = dtw[((size_t)g*DI + d)*RR + 0];
    float w1 = dtw[((size_t)g*DI + d)*RR + 1];
    float w2 = dtw[((size_t)g*DI + d)*RR + 2];
    float w3 = dtw[((size_t)g*DI + d)*RR + 3];
    float dtbv = dtb[(size_t)g*DI + d];
    float Dd   = Dp[(size_t)g*DI + d];
    float Av[NST], h[NST];
    #pragma unroll
    for (int n = 0; n < NST; ++n) {
        Av[n] = -__expf(A_log[((size_t)g*DI + d)*NST + n]);
        h[n]  = 0.f;
    }

    const size_t rowBase = ((size_t)(g*BB + b)) * LL;
    for (int t = 0; t < LL; ++t) {
        const float* row = dbl + (rowBase + t) * DBLP;
        float din = row[0]*w0 + row[1]*w1 + row[2]*w2 + row[3]*w3 + dtbv;
        float dt  = (din > 20.f) ? din : log1pf(__expf(din));
        float u   = uc[(rowBase + t)*DI + d];
        float du  = dt * u;
        float yv  = 0.f;
        #pragma unroll
        for (int n = 0; n < NST; ++n) {
            float Bn = row[4 + n], Cn = row[20 + n];
            h[n] = h[n] * __expf(dt * Av[n]) + du * Bn;
            yv  += h[n] * Cn;
        }
        int orig = dir ? (LL - 1 - t) : t;
        float z  = xz[(rowBase + orig)*(2*DI) + DI + d];
        float outv = (yv + u * Dd) * siluf(z);
        size_t oi = (rowBase + orig)*DI + d;
        if (dir) ySum[oi] += outv; else ySum[oi] = outv;
    }
}

// ---------------- gate combine: xo = g*xm + (1-g)*tok ----------------
__global__ void combine_kernel(const float* __restrict__ gate,
                               const float* __restrict__ xm,
                               const float* __restrict__ tok,
                               float* __restrict__ out, int n)
{
    int i = blockIdx.x * 256 + threadIdx.x;
    if (i < n) { float gv = gate[i]; out[i] = gv * xm[i] + (1.f - gv) * tok[i]; }
}

// ---------------- residual add ----------------
__global__ void add_kernel(const float* __restrict__ a, const float* __restrict__ b,
                           float* __restrict__ out, int n)
{
    int i = blockIdx.x * 256 + threadIdx.x;
    if (i < n) out[i] = a[i] + b[i];
}

// ---------------- host-side param plumbing ----------------
struct DirP  { const float *conv_w, *conv_b, *xw, *dtw, *dtb, *A_log, *D; };
struct LayerP {
    const float *ln_g, *ln_b, *pos, *gate_w, *gate_b, *proj_w, *proj_b,
                *in_w, *out_w, *out_b;
    DirP fwd, bwd;
};

static int fill_dir(void* const* d_in, int i, DirP& d) {
    d.conv_w = (const float*)d_in[i++]; d.conv_b = (const float*)d_in[i++];
    d.xw     = (const float*)d_in[i++]; d.dtw    = (const float*)d_in[i++];
    d.dtb    = (const float*)d_in[i++]; d.A_log  = (const float*)d_in[i++];
    d.D      = (const float*)d_in[i++];
    return i;
}
static int fill_layer(void* const* d_in, int i, LayerP& p) {
    p.ln_g   = (const float*)d_in[i++]; p.ln_b   = (const float*)d_in[i++];
    p.pos    = (const float*)d_in[i++]; p.gate_w = (const float*)d_in[i++];
    p.gate_b = (const float*)d_in[i++]; p.proj_w = (const float*)d_in[i++];
    p.proj_b = (const float*)d_in[i++]; p.in_w   = (const float*)d_in[i++];
    p.out_w  = (const float*)d_in[i++]; p.out_b  = (const float*)d_in[i++];
    i = fill_dir(d_in, i, p.fwd);
    i = fill_dir(d_in, i, p.bwd);
    return i;
}

// workspace layout (floats)
static constexpr size_t SZ_BCL  = (size_t)BB * CC * LL;        // 2359296
static constexpr size_t SZ_XZ   = (size_t)GG * MROWS * 256;    // 9437184
static constexpr size_t SZ_UC   = (size_t)GG * MROWS * DI;     // 4718592
static constexpr size_t SZ_DBL  = (size_t)GG * MROWS * DBLP;   // 1769472
static constexpr size_t OFF_HN   = 0;
static constexpr size_t OFF_TOK  = OFF_HN   + SZ_BCL;
static constexpr size_t OFF_XN   = OFF_TOK  + SZ_BCL;
static constexpr size_t OFF_GATE = OFF_XN   + SZ_BCL;
static constexpr size_t OFF_XZ   = OFF_GATE + SZ_BCL;
static constexpr size_t OFF_UCF  = OFF_XZ   + SZ_XZ;
static constexpr size_t OFF_UCB  = OFF_UCF  + SZ_UC;
static constexpr size_t OFF_DBLF = OFF_UCB  + SZ_UC;
static constexpr size_t OFF_DBLB = OFF_DBLF + SZ_DBL;
static constexpr size_t OFF_YSUM = OFF_DBLB + SZ_DBL;
static constexpr size_t OFF_XM   = OFF_YSUM + SZ_UC;
static constexpr size_t OFF_H1   = OFF_XM   + SZ_BCL;
static constexpr size_t OFF_H2   = OFF_H1   + SZ_BCL;

static void run_layer(hipStream_t stream, const float* hin, const LayerP& P,
                      float* ws, float* hout)
{
    float* hn     = ws + OFF_HN;
    float* tokens = ws + OFF_TOK;
    float* xn     = ws + OFF_XN;
    float* gate   = ws + OFF_GATE;
    float* xz     = ws + OFF_XZ;
    float* ucF    = ws + OFF_UCF;
    float* ucB    = ws + OFF_UCB;
    float* dblF   = ws + OFF_DBLF;
    float* dblB   = ws + OFF_DBLB;
    float* ySum   = ws + OFF_YSUM;
    float* xmamba = ws + OFF_XM;

    inorm_relu_kernel<<<BB*CC, 256, 0, stream>>>(hin, hn);
    tokens_ln_kernel<<<BB*LL, 256, 0, stream>>>(hn, P.pos, P.ln_g, P.ln_b, tokens, xn);

    dim3 gFull((CC + 63)/64, MROWS/32);
    // gate = sigmoid(xn @ gate_w^T + gate_b)
    gemm_wmma_bf16<<<gFull, 256, 0, stream>>>(xn, CC, 0, P.gate_w, P.gate_b,
                                              gate, CC, 0, CC, CC, 1, 0);
    // xz[g] = xn[:, g*Dg:(g+1)*Dg] @ in_w[g]^T
    for (int g = 0; g < GG; ++g)
        gemm_wmma_bf16<<<gFull, 256, 0, stream>>>(xn, CC, g*DG,
            P.in_w + (size_t)g*2*DI*DG, nullptr,
            xz + (size_t)g*MROWS*256, 256, 0, 2*DI, DG, 0, 0);

    int convBlocks = (GG*MROWS*DI)/256;
    conv_silu_kernel<<<convBlocks, 256, 0, stream>>>(xz, P.fwd.conv_w, P.fwd.conv_b, ucF, 0);
    conv_silu_kernel<<<convBlocks, 256, 0, stream>>>(xz, P.bwd.conv_w, P.bwd.conv_b, ucB, 1);

    dim3 gDbl(1, MROWS/32);
    for (int g = 0; g < GG; ++g) {
        gemm_wmma_bf16<<<gDbl, 256, 0, stream>>>(ucF + (size_t)g*MROWS*DI, DI, 0,
            P.fwd.xw + (size_t)g*DBLN*DI, nullptr,
            dblF + (size_t)g*MROWS*DBLP, DBLP, 0, DBLN, DI, 0, 0);
        gemm_wmma_bf16<<<gDbl, 256, 0, stream>>>(ucB + (size_t)g*MROWS*DI, DI, 0,
            P.bwd.xw + (size_t)g*DBLN*DI, nullptr,
            dblB + (size_t)g*MROWS*DBLP, DBLP, 0, DBLN, DI, 0, 0);
    }

    scan_kernel<<<GG*BB, DI, 0, stream>>>(ucF, dblF, xz, P.fwd.dtw, P.fwd.dtb,
                                          P.fwd.A_log, P.fwd.D, ySum, 0);
    scan_kernel<<<GG*BB, DI, 0, stream>>>(ucB, dblB, xz, P.bwd.dtw, P.bwd.dtb,
                                          P.bwd.A_log, P.bwd.D, ySum, 1);

    dim3 gOut((DG + 63)/64, MROWS/32);
    for (int g = 0; g < GG; ++g)
        gemm_wmma_bf16<<<gOut, 256, 0, stream>>>(ySum + (size_t)g*MROWS*DI, DI, 0,
            P.out_w + (size_t)g*DG*DI, P.out_b + (size_t)g*DG,
            xmamba, CC, g*DG, DG, DI, 0, 0);

    combine_kernel<<<(int)(SZ_BCL/256), 256, 0, stream>>>(gate, xmamba, tokens, xn, (int)SZ_BCL);

    // proj with transposed store -> hout in (B,C,L)
    gemm_wmma_bf16<<<gFull, 256, 0, stream>>>(xn, CC, 0, P.proj_w, P.proj_b,
                                              hout, 0, 0, CC, CC, 0, 1);
}

extern "C" void kernel_launch(void* const* d_in, const int* in_sizes, int n_in,
                              void* d_out, int out_size, void* d_ws, size_t ws_size,
                              hipStream_t stream)
{
    (void)in_sizes; (void)n_in; (void)out_size; (void)ws_size;
    const float* x = (const float*)d_in[0];
    LayerP p1, p2;
    int i = 1;
    i = fill_layer(d_in, i, p1);
    i = fill_layer(d_in, i, p2);

    float* ws = (float*)d_ws;
    run_layer(stream, x,            p1, ws, ws + OFF_H1);
    run_layer(stream, ws + OFF_H1,  p2, ws, ws + OFF_H2);
    add_kernel<<<(int)(SZ_BCL/256), 256, 0, stream>>>(ws + OFF_H2, x,
                                                      (float*)d_out, (int)SZ_BCL);
}
